// DecoupledMLPGaussianActor_82274393522350
// MI455X (gfx1250) — matrix operand order repaired and checked by reading, not compiled, not run on hardware
//
#include <hip/hip_runtime.h>
#include <hip/hip_bf16.h>
#include <math.h>

typedef __attribute__((ext_vector_type(16))) _Float16 v16h;
typedef __attribute__((ext_vector_type(8)))  _Float16 v8h;
typedef __attribute__((ext_vector_type(2)))  _Float16 v2h;
typedef __attribute__((ext_vector_type(8)))  float    v8f;
typedef __attribute__((ext_vector_type(2)))  float    v2f;

#define OBS_DIM 512
#define ACT_DIM 32
#define H1 1024
#define H2 1024
#define BATCH 4096
#define BT 64
#define LOG2PI 1.8378770664093453f

// CDNA5 hardware tanh (TRANS32 op). One v_nop satisfies the trans-op result
// hazard (ISA: 1 independent op or V_NOP before output is consumed); the
// compiler cannot insert it across an asm boundary, so we do.
__device__ __forceinline__ float htanh(float x) {
  float r;
  asm volatile("v_tanh_f32 %0, %1\n\tv_nop" : "=v"(r) : "v"(x));
  return r;
}

// ---------------------------------------------------------------------------
// Weight conversion + swizzle: f32 row-major [A, K, N]  ->  f16 fragment-major
// WMMA B-operand layout. Fragment = 32x16 (KxN) tile = 512 halves; lane L
// holds 16 contiguous halves: k = k0 + (L>>4)*16 + j, n = n0 + (L&15).
// Fragments ordered k-inner so the GEMM K-loop streams contiguously.
// ---------------------------------------------------------------------------
__global__ void k_swizzle(const float* __restrict__ src, _Float16* __restrict__ dst,
                          int K, int N, long long total) {
  long long gid = (long long)blockIdx.x * blockDim.x + threadIdx.x;
  if (gid >= total) return;
  int perHead = K * N;
  int a = (int)(gid / perHead);
  int e = (int)(gid % perHead);
  int f = e >> 9;           // fragment index (512 halves each)
  int r = e & 511;
  int L = r >> 4;           // lane 0..31
  int j = r & 15;           // half index within lane
  int ktiles = K >> 5;
  int k0 = (f % ktiles) << 5;
  int n0 = (f / ktiles) << 4;
  int n  = n0 + (L & 15);
  int k  = k0 + ((L >> 4) << 4) + j;
  dst[gid] = (_Float16)src[(long long)a * perHead + (long long)k * N + n];
}

// ---------------------------------------------------------------------------
// Fused 3-layer actor MLP per (batch-tile, head).
// 256 threads = 8 wave32. Wave w: rows m0 = (w>>1)*16, N-half = (w&1)*512.
// obs tile converted to f16 in LDS exactly once; x1 lives only in LDS (f16);
// x2 never materializes (layer-3 fused into the layer-2 epilogue). mu reduced
// deterministically (shfl + LDS staging).  LDS: 64 + 128 KB < 320 KB/WGP.
// ---------------------------------------------------------------------------
__launch_bounds__(256, 1)
__global__ void k_actor(const float* __restrict__ obs,
                        const _Float16* __restrict__ W1h,
                        const float* __restrict__ b1,
                        const _Float16* __restrict__ W2h,
                        const float* __restrict__ b2,
                        const float* __restrict__ W3,
                        const float* __restrict__ b3,
                        float* __restrict__ muOut) {
  __shared__ _Float16 sObs[BT * OBS_DIM];  //  64 KB: obs tile, f16 row-major
  __shared__ _Float16 sX1[BT * H1];        // 128 KB: x1 tile, f16 row-major
  __shared__ float    sRed[8][16];         // per-wave mu partials

  const int tile = blockIdx.x;
  const int a    = blockIdx.y;
  const int tid  = threadIdx.x;
  const int wave = tid >> 5;
  const int lane = tid & 31;
  const int half = lane >> 4;
  const int lq   = lane & 15;
  const int m0   = (wave >> 1) << 4;       // 0,16,32,48
  const int nbase = (wave & 1) * 512;
  const int rowg = tile * BT;

  const _Float16* Wa1 = W1h + (size_t)a * OBS_DIM * H1;
  const _Float16* Wa2 = W2h + (size_t)a * H1 * H2;

  // ---- stage obs tile to LDS as f16 (each element converted exactly once) ----
  {
    const v2f* src = (const v2f*)(obs + (size_t)rowg * OBS_DIM);
#pragma unroll
    for (int i = 0; i < (BT * OBS_DIM) / (256 * 2); ++i) {
      int e = tid + i * 256;               // float2-granule index, coalesced
      v2f p = src[e];
      v2h h;
      h[0] = (_Float16)p.x;
      h[1] = (_Float16)p.y;
      *(v2h*)&sObs[e * 2] = h;             // v_cvt_pk_f16_f32 + ds_store_b32
    }
  }
  __syncthreads();

  // ---------------- layer 1: x1 = tanh(obs @ W1_a + b1_a) ----------------
  const _Float16* orow = sObs + (size_t)(m0 + lq) * OBS_DIM;
  for (int nt = 0; nt < 32; ++nt) {
    const int n0 = nbase + (nt << 4);
    v8f acc = {};
    for (int k0 = 0; k0 < OBS_DIM; k0 += 32) {
      v8h lo = *(const v8h*)(orow + k0 + half * 16);
      v8h hi = *(const v8h*)(orow + k0 + half * 16 + 8);
      v16h A = __builtin_shufflevector(lo, hi, 0, 1, 2, 3, 4, 5, 6, 7,
                                       8, 9, 10, 11, 12, 13, 14, 15);
      // B fragment: one coalesced 32B load per lane from pre-swizzled W1h
      const int f = ((n0 >> 4) * (OBS_DIM >> 5)) + (k0 >> 5);
      const _Float16* bp = Wa1 + (size_t)f * 512 + lane * 16;
      __builtin_prefetch(bp + 2 * 512, 0, 1);   // global_prefetch_b8 ahead
      v16h B = *(const v16h*)bp;
      acc = __builtin_amdgcn_wmma_f32_16x16x32_f16(false, A, false, B,
                                                   (short)0, acc, false, false);
    }
    const int n = n0 + lq;
    const float bias = b1[a * H1 + n];
#pragma unroll
    for (int rr = 0; rr < 8; ++rr) {
      float v = htanh(acc[rr] + bias);
      sX1[(m0 + rr + half * 8) * H1 + n] = (_Float16)v;   // C-layout scatter
    }
  }
  __syncthreads();

  // ------------- layer 2 + 3: mu = tanh(x1 @ W2_a + b2_a) . w3_a -------------
  float mup[8];
#pragma unroll
  for (int rr = 0; rr < 8; ++rr) mup[rr] = 0.0f;

  const _Float16* xrow = sX1 + (size_t)(m0 + lq) * H1;
  for (int nt = 0; nt < 32; ++nt) {
    const int n0 = nbase + (nt << 4);
    v8f acc = {};
    for (int k0 = 0; k0 < H1; k0 += 32) {
      v8h lo = *(const v8h*)(xrow + k0 + half * 16);
      v8h hi = *(const v8h*)(xrow + k0 + half * 16 + 8);
      v16h A = __builtin_shufflevector(lo, hi, 0, 1, 2, 3, 4, 5, 6, 7,
                                       8, 9, 10, 11, 12, 13, 14, 15);
      const int f = ((n0 >> 4) * (H1 >> 5)) + (k0 >> 5);
      const _Float16* bp = Wa2 + (size_t)f * 512 + lane * 16;
      __builtin_prefetch(bp + 2 * 512, 0, 1);
      v16h B = *(const v16h*)bp;
      acc = __builtin_amdgcn_wmma_f32_16x16x32_f16(false, A, false, B,
                                                   (short)0, acc, false, false);
    }
    const int n = n0 + lq;
    const float bias = b2[a * H2 + n];
    const float w3v  = W3[a * H2 + n];
#pragma unroll
    for (int rr = 0; rr < 8; ++rr) {
      float v = htanh(acc[rr] + bias);
      mup[rr] += v * w3v;                 // layer-3 fused: x2 never stored
    }
  }

  // deterministic reduction over the 16 column-lanes of each half
#pragma unroll
  for (int rr = 0; rr < 8; ++rr) {
#pragma unroll
    for (int off = 1; off < 16; off <<= 1)
      mup[rr] += __shfl_xor(mup[rr], off, 32);
  }
  if (lq == 0) {
#pragma unroll
    for (int rr = 0; rr < 8; ++rr)
      sRed[wave][half * 8 + rr] = mup[rr];
  }
  __syncthreads();

  if (tid < BT) {
    const int row = tid;
    const int mt = row >> 4;
    const int r8 = row & 15;
    float mu = sRed[2 * mt][r8] + sRed[2 * mt + 1][r8] + b3[a];
    muOut[(size_t)(rowg + row) * ACT_DIM + a] = mu;
  }
}

// ---------------------------------------------------------------------------
// logp[b] = sum_a ( -0.5*(z^2 + log(2pi)) - log_std[a] ),
// z = (act - mu) * exp(-log_std)
// ---------------------------------------------------------------------------
__global__ void k_logp(const float* __restrict__ act, const float* __restrict__ mu,
                       const float* __restrict__ log_std, float* __restrict__ logp) {
  int b = blockIdx.x * blockDim.x + threadIdx.x;
  if (b >= BATCH) return;
  float s = 0.0f;
#pragma unroll 8
  for (int a = 0; a < ACT_DIM; ++a) {
    float ls = log_std[a];
    float z = (act[b * ACT_DIM + a] - mu[b * ACT_DIM + a]) * __expf(-ls);
    s += -0.5f * (z * z + LOG2PI) - ls;
  }
  logp[b] = s;
}

extern "C" void kernel_launch(void* const* d_in, const int* in_sizes, int n_in,
                              void* d_out, int out_size, void* d_ws, size_t ws_size,
                              hipStream_t stream) {
  const float* obs = (const float*)d_in[0];
  const float* act = (const float*)d_in[1];
  const float* W1  = (const float*)d_in[2];
  const float* b1  = (const float*)d_in[3];
  const float* W2  = (const float*)d_in[4];
  const float* b2  = (const float*)d_in[5];
  const float* W3  = (const float*)d_in[6];
  const float* b3  = (const float*)d_in[7];
  const float* lsd = (const float*)d_in[8];

  _Float16* W1h = (_Float16*)d_ws;                              // 33.5 MB
  _Float16* W2h = W1h + (size_t)ACT_DIM * OBS_DIM * H1;         // 67 MB

  long long t1 = (long long)ACT_DIM * OBS_DIM * H1;
  long long t2 = (long long)ACT_DIM * H1 * H2;
  k_swizzle<<<dim3((unsigned)((t1 + 255) / 256)), 256, 0, stream>>>(W1, W1h, OBS_DIM, H1, t1);
  k_swizzle<<<dim3((unsigned)((t2 + 255) / 256)), 256, 0, stream>>>(W2, W2h, H1, H2, t2);

  float* mu   = (float*)d_out;
  float* logp = mu + (size_t)BATCH * ACT_DIM;
  k_actor<<<dim3(BATCH / BT, ACT_DIM), 256, 0, stream>>>(obs, W1h, b1, W2h, b2, W3, b3, mu);
  k_logp<<<dim3(BATCH / 256), 256, 0, stream>>>(act, mu, lsd, logp);
}
